// Model_25451976196926
// MI455X (gfx1250) — matrix operand order repaired and checked by reading, not compiled, not run on hardware
//
#include <hip/hip_runtime.h>

typedef __bf16 bf16;
typedef __attribute__((ext_vector_type(16))) __bf16 v16bf;
typedef __attribute__((ext_vector_type(8)))  __bf16 v8bf;
typedef __attribute__((ext_vector_type(8)))  float  v8f;
typedef __attribute__((ext_vector_type(4)))  unsigned int v4u;
typedef __attribute__((ext_vector_type(8)))  int v8i;
typedef __attribute__((ext_vector_type(4)))  int v4i;

#define N_LAYER  12
#define D_MODEL  512
#define N_HEAD   8
#define D_HEAD   64
#define D_FF     2048
#define N_TOKEN  332
#define MEM_LEN  512
#define BATCH    4
#define QLEN     512
#define KLEN     1024   // QLEN + MEM_LEN

#define MT 4            // 16-row tiles per wave (M)
#define NT 2            // 16-col tiles per wave (N)
#define KC 64           // K-chunk staged to LDS per TDM transfer (divides all Kd)

// ---------------------------------------------------------------------------
// TDM: issue a 2D tensor_load_to_lds of tile [tile_d1 rows x tile_d0 elems]
// of 2-byte elements from a row-major tensor (row stride d0_stride elems).
// Rows >= tensor_d1 zero-fill (OOB read -> 0). D# per CDNA5 ISA 8.3/8.4.
// ---------------------------------------------------------------------------
__device__ inline void tdm_load_2d_b16(unsigned lds_byte_addr, const void* gaddr,
                                       unsigned tile_d0, unsigned tile_d1,
                                       unsigned tensor_d0, unsigned tensor_d1,
                                       unsigned long long d0_stride)
{
    unsigned long long ga = (unsigned long long)(size_t)gaddr;
    v4u g0;
    g0[0] = 1u;                                        // count=1 (valid user D#)
    g0[1] = lds_byte_addr;                             // lds_addr [63:32]
    g0[2] = (unsigned)(ga & 0xFFFFFFFFu);              // global_addr [95:64]
    g0[3] = (unsigned)((ga >> 32) & 0x01FFFFFFu)       // global_addr [120:96]
          | (2u << 30);                                // type=2 ("image") [127:126]
    v8i g1;
    g1[0] = (int)(1u << 16);                           // data_size=1 (2B); wg_mask=0
    g1[1] = (int)((tensor_d0 & 0xFFFFu) << 16);        // tensor_dim0 lo16 (atomic_addr=0)
    g1[2] = (int)(((tensor_d0 >> 16) & 0xFFFFu) | ((tensor_d1 & 0xFFFFu) << 16));
    g1[3] = (int)(((tensor_d1 >> 16) & 0xFFFFu) | ((tile_d0 & 0xFFFFu) << 16));
    g1[4] = (int)(tile_d1 & 0xFFFFu);                  // tile_dim1 | tile_dim2=0
    g1[5] = (int)(unsigned)(d0_stride & 0xFFFFFFFFu);  // tensor_dim0_stride lo32
    g1[6] = (int)(unsigned)((d0_stride >> 32) & 0xFFFFu); // stride hi16 | dim1_stride=0
    g1[7] = 0;
    v4i z4; z4[0] = z4[1] = z4[2] = z4[3] = 0;
#if defined(__clang_major__) && (__clang_major__ >= 23)
    v8i z8;
#pragma unroll
    for (int t = 0; t < 8; ++t) z8[t] = 0;
    __builtin_amdgcn_tensor_load_to_lds(g0, g1, z4, z4, z8, 0);
#else
    __builtin_amdgcn_tensor_load_to_lds(g0, g1, z4, z4, 0);
#endif
}

// ---------------------------------------------------------------------------
// Register-blocked WMMA bf16 GEMM: C[m,n] = alpha*sum_k A[m,k]*W[n,k] (+bias,+resid,relu)
// Each wave computes 64x32 C (4x2 tiles) via v_wmma_f32_16x16x32_bf16.
// Weight slab (32 cols x KC) staged to LDS once per block by the Tensor Data
// Mover, double-buffered (TENSORcnt); A streams from global with prefetch.
// Batch (grid.z): offsets decomposed as z1=z/8, z2=z%8 -> off = z1*S1 + z2*S2.
// ---------------------------------------------------------------------------
__global__ __launch_bounds__(128) void wmma_gemm_bf16(
    const bf16* __restrict__ A, long long aS1, long long aS2, int lda,
    const bf16* __restrict__ W, long long wS1, long long wS2, int ldw,
    float* __restrict__ outF, bf16* __restrict__ outB,
    long long cS1, long long cS2, int ldc,
    const float* __restrict__ bias, const float* __restrict__ resid,
    float alpha, int M, int N, int Kd, int relu)
{
    __shared__ bf16 sB[2][(NT * 16) * KC];             // 2 x 4KB weight slabs

    int z  = blockIdx.z;
    long long z1 = z >> 3, z2 = z & 7;
    A += z1 * aS1 + z2 * aS2;
    W += z1 * wS1 + z2 * wS2;
    long long coff = z1 * cS1 + z2 * cS2;

    int lane   = threadIdx.x;                          // 0..31, wave32
    int tileN0 = blockIdx.x * (NT * 16);
    int tileM0 = blockIdx.y * (4 * MT * 16) + threadIdx.y * (MT * 16);
    // (all launch shapes keep every wave in-bounds; no early exit before barriers)

    bool hi  = lane >= 16;
    int  l16 = lane & 15;

    // A fragments (16x32): lane row m=l16; lanes<16 K {0..7,16..23}, lanes>=16 K {8..15,24..31}
    const bf16* ap[MT];
#pragma unroll
    for (int ti = 0; ti < MT; ++ti) {
        int m = tileM0 + ti * 16 + l16;
        int mA = (m < M) ? m : (M - 1);
        ap[ti] = A + (size_t)mA * lda + (hi ? 8 : 0);
    }
    // B fragments come from the LDS slab: row n_local, K {0..7,16..23} / {8..15,24..31}.
    const bf16* sp[NT];
#pragma unroll
    for (int tj = 0; tj < NT; ++tj)
        sp[tj] = &sB[0][0] + (tj * 16 + l16) * KC + (hi ? 8 : 0);

    const bf16* Wt = W + (size_t)tileN0 * ldw;         // weight tile origin (row tileN0)
    unsigned tensRows = (unsigned)(N - tileN0);        // rows available; rest zero-fill

    v8f acc[MT][NT];
#pragma unroll
    for (int ti = 0; ti < MT; ++ti)
#pragma unroll
        for (int tj = 0; tj < NT; ++tj)
#pragma unroll
            for (int t = 0; t < 8; ++t) acc[ti][tj][t] = 0.0f;

    int nchunks = Kd / KC;
    // prime the pipeline: DMA chunk 0 into buffer 0
    if (threadIdx.y == 0) {
        tdm_load_2d_b16((unsigned)(size_t)&sB[0][0], Wt,
                        KC, NT * 16, KC, tensRows, (unsigned long long)ldw);
    }

    for (int c = 0; c < nchunks; ++c) {
        if (threadIdx.y == 0) {
            if (c + 1 < nchunks) {                     // DMA next chunk, keep 1 in flight
                tdm_load_2d_b16((unsigned)(size_t)&sB[(c + 1) & 1][0],
                                Wt + (size_t)(c + 1) * KC,
                                KC, NT * 16, KC, tensRows, (unsigned long long)ldw);
                __builtin_amdgcn_s_wait_tensorcnt(1);  // chunk c landed
            } else {
                __builtin_amdgcn_s_wait_tensorcnt(0);  // drain
            }
        }
        __syncthreads();                               // publish slab c to all waves

        const bf16* bb = &sB[c & 1][0] - &sB[0][0] + sp[0];      // keep sp math simple
        (void)bb;
#pragma unroll
        for (int kk = 0; kk < KC; kk += 32) {
            int k = c * KC + kk;
            v16bf bv[NT];
#pragma unroll
            for (int tj = 0; tj < NT; ++tj) {
                v8bf b0 = *(const v8bf*)(sp[tj] + ((c & 1) ? (NT * 16) * KC : 0) + kk);
                v8bf b1 = *(const v8bf*)(sp[tj] + ((c & 1) ? (NT * 16) * KC : 0) + kk + 16);
#pragma unroll
                for (int t = 0; t < 8; ++t) { bv[tj][t] = b0[t]; bv[tj][8 + t] = b1[t]; }
            }
            v16bf av[MT];
#pragma unroll
            for (int ti = 0; ti < MT; ++ti) {
                v8bf a0 = *(const v8bf*)(ap[ti] + k);
                v8bf a1 = *(const v8bf*)(ap[ti] + k + 16);
#pragma unroll
                for (int t = 0; t < 8; ++t) { av[ti][t] = a0[t]; av[ti][8 + t] = a1[t]; }
            }
#pragma unroll
            for (int ti = 0; ti < MT; ++ti)
#pragma unroll
                for (int tj = 0; tj < NT; ++tj)
                    acc[ti][tj] = __builtin_amdgcn_wmma_f32_16x16x32_bf16(
                        false, av[ti], false, bv[tj], (short)0, acc[ti][tj], false, false);
        }
        // speculative prefetch of the A stream one chunk ahead (OOB dropped)
        __builtin_prefetch(ap[0] + c * KC + KC, 0, 1);
        __syncthreads();                               // slab c consumed; safe to overwrite
    }

    // C layout: lane col n = l16; VGPR r -> row m = tile + r + (hi?8:0)
#pragma unroll
    for (int tj = 0; tj < NT; ++tj) {
        int n = tileN0 + tj * 16 + l16;
        int nW = (n < N) ? n : (N - 1);
        float bn = bias ? bias[nW] : 0.0f;
#pragma unroll
        for (int ti = 0; ti < MT; ++ti) {
#pragma unroll
            for (int r = 0; r < 8; ++r) {
                int mm = tileM0 + ti * 16 + r + (hi ? 8 : 0);
                if (mm < M && n < N) {
                    size_t ci = (size_t)coff + (size_t)mm * ldc + n;
                    float v = acc[ti][tj][r] * alpha + bn;
                    if (resid) v += resid[ci];
                    if (relu)  v = fmaxf(v, 0.0f);
                    if (outF) outF[ci] = v;
                    if (outB) outB[ci] = (bf16)v;
                }
            }
        }
    }
}

// ---------------------------------------------------------------------------
// LayerNorm over rows of length 512 -> bf16 output.
// catMode: rows are [B,KLEN]; mem rows (j<512) are LN(0) = b; others from h[b, j-512].
// ---------------------------------------------------------------------------
__global__ __launch_bounds__(256) void ln_kernel(
    const float* __restrict__ x, const float* __restrict__ g,
    const float* __restrict__ b, bf16* __restrict__ out, int catMode)
{
    __shared__ float sh[256];
    int row = blockIdx.x;
    int t   = threadIdx.x;
    bf16* orow = out + (size_t)row * D_MODEL;
    const float* xr;
    if (catMode) {
        int bb = row >> 10, j = row & (KLEN - 1);
        if (j < MEM_LEN) {                      // LN of zeros -> bias (uniform branch)
            orow[t]       = (bf16)b[t];
            orow[t + 256] = (bf16)b[t + 256];
            return;
        }
        xr = x + ((size_t)bb * QLEN + (j - MEM_LEN)) * D_MODEL;
    } else {
        xr = x + (size_t)row * D_MODEL;
    }
    float x0 = xr[t], x1 = xr[t + 256];
    sh[t] = x0 + x1; __syncthreads();
    for (int s = 128; s > 0; s >>= 1) { if (t < s) sh[t] += sh[t + s]; __syncthreads(); }
    float mu = sh[0] * (1.0f / 512.0f); __syncthreads();
    float d0 = x0 - mu, d1 = x1 - mu;
    sh[t] = d0 * d0 + d1 * d1; __syncthreads();
    for (int s = 128; s > 0; s >>= 1) { if (t < s) sh[t] += sh[t + s]; __syncthreads(); }
    float inv = rsqrtf(sh[0] * (1.0f / 512.0f) + 1e-3f);
    orow[t]       = (bf16)(d0 * inv * g[t] + b[t]);
    orow[t + 256] = (bf16)(d1 * inv * g[t + 256] + b[t + 256]);
}

// ---------------------------------------------------------------------------
// Split qkv [B*KLEN, 1536] (f32) into per-head bf16 tensors:
//  Qw/Qr [B,H,Q,64] (q + r_w_bias / r_r_bias, only rows j>=512)
//  Kh    [B,H,K,64]
//  Vt    [B,H,64,K]  (transposed for prob@V via A*W^T kernel)
// ---------------------------------------------------------------------------
__global__ __launch_bounds__(256) void split_qkv_kernel(
    const float* __restrict__ qkv,
    const float* __restrict__ rwb, const float* __restrict__ rrb,
    bf16* __restrict__ Qw, bf16* __restrict__ Qr,
    bf16* __restrict__ Kh, bf16* __restrict__ Vt)
{
    int row = blockIdx.x;                  // 0..B*KLEN-1
    int bb = row >> 10, j = row & (KLEN - 1);
    const float* src = qkv + (size_t)row * (3 * D_MODEL);
    int t = threadIdx.x;
#pragma unroll
    for (int it = 0; it < 2; ++it) {
        int d  = t + it * 256;
        int hh = d >> 6, dd = d & 63;
        size_t bh = (size_t)(bb * N_HEAD + hh);
        Kh[(bh * KLEN + j) * D_HEAD + dd] = (bf16)src[D_MODEL + d];
        Vt[(bh * D_HEAD + dd) * KLEN + j] = (bf16)src[2 * D_MODEL + d];
        if (j >= MEM_LEN) {
            int qq = j - MEM_LEN;
            float qv = src[d];
            Qw[(bh * QLEN + qq) * D_HEAD + dd] = (bf16)(qv + rwb[d]);
            Qr[(bh * QLEN + qq) * D_HEAD + dd] = (bf16)(qv + rrb[d]);
        }
    }
}

// rk f32 [K, 512] -> bf16 [H, K, 64]
__global__ __launch_bounds__(256) void rk_permute_kernel(
    const float* __restrict__ rkC, bf16* __restrict__ rkB)
{
    int k = blockIdx.x;  int t = threadIdx.x;
#pragma unroll
    for (int it = 0; it < 2; ++it) {
        int d = t + it * 256;
        int hh = d >> 6, dd = d & 63;
        rkB[((size_t)hh * KLEN + k) * D_HEAD + dd] = (bf16)rkC[(size_t)k * D_MODEL + d];
    }
}

// Sinusoidal relative position embedding, pos descending K-1..0 -> bf16 [K,512]
__global__ __launch_bounds__(256) void posemb_kernel(bf16* __restrict__ out)
{
    int k = blockIdx.x;  int t = threadIdx.x;
    float pos = (float)(KLEN - 1 - k);
#pragma unroll
    for (int it = 0; it < 2; ++it) {
        int i  = t + it * 256;
        int fi = (i < 256) ? i : (i - 256);
        float fr = __expf(-(float)fi * (9.210340371976184f / 256.0f)); // 10000^(-fi/256)
        float s  = pos * fr;
        out[(size_t)k * D_MODEL + i] = (bf16)((i < 256) ? __sinf(s) : __cosf(s));
    }
}

// Embedding gather -> bf16 rows for proj GEMM
__global__ __launch_bounds__(256) void gather_kernel(
    const int* __restrict__ ids, const float* __restrict__ emb, bf16* __restrict__ out)
{
    int row = blockIdx.x;  int t = threadIdx.x;
    const float* e = emb + (size_t)ids[row] * D_MODEL;
    out[(size_t)row * D_MODEL + t]       = (bf16)e[t];
    out[(size_t)row * D_MODEL + t + 256] = (bf16)e[t + 256];
}

__global__ __launch_bounds__(256) void f32_to_bf16_kernel(
    const float* __restrict__ in, bf16* __restrict__ out, long long n)
{
    long long i = (long long)blockIdx.x * 256 + threadIdx.x;
    long long stride = (long long)gridDim.x * 256;
    for (; i < n; i += stride) out[i] = (bf16)in[i];
}

// ---------------------------------------------------------------------------
// Fused rel-shift + causal mask + softmax.
// score[q,j] = (AC[q,j] + BD[q, j-q+511]) / 8  for j <= q+512, else -inf.
// One block per (b,h,q) row of length KLEN.
// ---------------------------------------------------------------------------
__global__ __launch_bounds__(256) void attn_softmax_kernel(
    const float* __restrict__ AC, const float* __restrict__ BD, bf16* __restrict__ prob)
{
    __shared__ float sh[256];
    int idx = blockIdx.x;                   // z*QLEN + q
    int q   = idx & (QLEN - 1);
    const float* ac = AC + (size_t)idx * KLEN;
    const float* bd = BD + (size_t)idx * KLEN;
    bf16* p = prob + (size_t)idx * KLEN;
    int t = threadIdx.x;
    int lim = q + MEM_LEN;                  // allowed: j <= lim  (then j-q+511 in [0,1023])
    float vals[4];
    float mx = -3.0e38f;
#pragma unroll
    for (int it = 0; it < 4; ++it) {
        int j = it * 256 + t;
        float v = (j <= lim) ? (ac[j] + bd[j - q + (MEM_LEN - 1)]) * 0.125f : -3.0e38f;
        vals[it] = v;
        mx = fmaxf(mx, v);
    }
    sh[t] = mx; __syncthreads();
    for (int s = 128; s > 0; s >>= 1) { if (t < s) sh[t] = fmaxf(sh[t], sh[t + s]); __syncthreads(); }
    float rowMax = sh[0]; __syncthreads();
    float sum = 0.0f;
#pragma unroll
    for (int it = 0; it < 4; ++it) {
        int j = it * 256 + t;
        float e = (j <= lim) ? __expf(vals[it] - rowMax) : 0.0f;
        vals[it] = e;
        sum += e;
    }
    sh[t] = sum; __syncthreads();
    for (int s = 128; s > 0; s >>= 1) { if (t < s) sh[t] += sh[t + s]; __syncthreads(); }
    float inv = 1.0f / sh[0];
#pragma unroll
    for (int it = 0; it < 4; ++it) {
        int j = it * 256 + t;
        p[j] = (bf16)(vals[it] * inv);
    }
}

// ---------------------------------------------------------------------------
extern "C" void kernel_launch(void* const* d_in, const int* in_sizes, int n_in,
                              void* d_out, int out_size, void* d_ws, size_t ws_size,
                              hipStream_t stream)
{
    (void)in_sizes; (void)n_in; (void)out_size; (void)ws_size;

    const int*   input_ids = (const int*)  d_in[0];
    const float* emb       = (const float*)d_in[1];
    const float* emb_proj  = (const float*)d_in[2];
    const float* r_w_bias  = (const float*)d_in[3];
    const float* r_r_bias  = (const float*)d_in[4];
    const float* qkv_w     = (const float*)d_in[5];
    const float* o_w       = (const float*)d_in[6];
    const float* r_net_w   = (const float*)d_in[7];
    const float* ln1_g     = (const float*)d_in[8];
    const float* ln1_b     = (const float*)d_in[9];
    const float* ff1_w     = (const float*)d_in[10];
    const float* ff1_b     = (const float*)d_in[11];
    const float* ff2_w     = (const float*)d_in[12];
    const float* ff2_b     = (const float*)d_in[13];
    const float* ln2_g     = (const float*)d_in[14];
    const float* ln2_b     = (const float*)d_in[15];
    const float* head_w    = (const float*)d_in[16];
    const float* head_b    = (const float*)d_in[17];
    float* outp = (float*)d_out;

    char* wsp = (char*)d_ws; size_t off = 0;
    auto alloc = [&](size_t bytes) -> void* {
        void* p = wsp + off;
        off += (bytes + 255) & ~(size_t)255;
        return p;
    };

    const int BQ  = BATCH * QLEN;   // 2048
    const int BK  = BATCH * KLEN;   // 4096
    const int BH  = BATCH * N_HEAD; // 32

    bf16*  epB    = (bf16*) alloc((size_t)D_MODEL * D_MODEL * 2);
    bf16*  posB   = (bf16*) alloc((size_t)KLEN * D_MODEL * 2);
    bf16*  gatB   = (bf16*) alloc((size_t)BQ * D_MODEL * 2);
    float* hF     = (float*)alloc((size_t)BQ * D_MODEL * 4);
    bf16*  headB  = (bf16*) alloc((size_t)N_TOKEN * D_MODEL * 2);
    bf16*  wqB    = (bf16*) alloc((size_t)3 * D_MODEL * D_MODEL * 2);
    bf16*  woB    = (bf16*) alloc((size_t)D_MODEL * D_MODEL * 2);
    bf16*  wrB    = (bf16*) alloc((size_t)D_MODEL * D_MODEL * 2);
    bf16*  wf1B   = (bf16*) alloc((size_t)D_FF * D_MODEL * 2);
    bf16*  wf2B   = (bf16*) alloc((size_t)D_MODEL * D_FF * 2);
    bf16*  XlnB   = (bf16*) alloc((size_t)BK * D_MODEL * 2);
    float* qkvC   = (float*)alloc((size_t)BK * 3 * D_MODEL * 4);
    bf16*  QwB    = (bf16*) alloc((size_t)BH * QLEN * D_HEAD * 2);
    bf16*  QrB    = (bf16*) alloc((size_t)BH * QLEN * D_HEAD * 2);
    bf16*  KhB    = (bf16*) alloc((size_t)BH * KLEN * D_HEAD * 2);
    bf16*  VtB    = (bf16*) alloc((size_t)BH * D_HEAD * KLEN * 2);
    float* rkC    = (float*)alloc((size_t)KLEN * D_MODEL * 4);
    bf16*  rkB    = (bf16*) alloc((size_t)N_HEAD * KLEN * D_HEAD * 2);
    float* ACf    = (float*)alloc((size_t)BH * QLEN * KLEN * 4);
    float* BDf    = (float*)alloc((size_t)BH * QLEN * KLEN * 4);
    bf16*  probB  = (bf16*) alloc((size_t)BH * QLEN * KLEN * 2);
    bf16*  attnB  = (bf16*) alloc((size_t)BQ * D_MODEL * 2);
    bf16*  X2B    = (bf16*) alloc((size_t)BQ * D_MODEL * 2);
    bf16*  XffB   = (bf16*) alloc((size_t)BQ * D_FF * 2);
    bf16*  hB     = (bf16*) alloc((size_t)BQ * D_MODEL * 2);

    auto conv = [&](const float* src, bf16* dst, long long n) {
        int g = (int)((n + 255) / 256); if (g > 4096) g = 4096;
        f32_to_bf16_kernel<<<g, 256, 0, stream>>>(src, dst, n);
    };
    auto gemm = [&](const bf16* A, long long aS1, long long aS2, int lda,
                    const bf16* W, long long wS1, long long wS2, int ldw,
                    float* oF, bf16* oB, long long cS1, long long cS2, int ldc,
                    const float* bias, const float* resid, float alpha,
                    int M, int N, int Kd, int relu, int batch) {
        dim3 blk(32, 4, 1);
        dim3 grd((N + NT * 16 - 1) / (NT * 16), (M + 4 * MT * 16 - 1) / (4 * MT * 16), batch);
        wmma_gemm_bf16<<<grd, blk, 0, stream>>>(A, aS1, aS2, lda, W, wS1, wS2, ldw,
                                                oF, oB, cS1, cS2, ldc,
                                                bias, resid, alpha, M, N, Kd, relu);
    };

    // ---- embedding: h = (emb[ids] @ emb_proj^T) * sqrt(D)
    conv(emb_proj, epB, (long long)D_MODEL * D_MODEL);
    conv(head_w,  headB, (long long)N_TOKEN * D_MODEL);
    posemb_kernel<<<KLEN, 256, 0, stream>>>(posB);
    gather_kernel<<<BQ, 256, 0, stream>>>(input_ids, emb, gatB);
    gemm(gatB, 0, 0, D_MODEL, epB, 0, 0, D_MODEL,
         hF, nullptr, 0, 0, D_MODEL, nullptr, nullptr,
         22.62741699796952f, BQ, D_MODEL, D_MODEL, 0, 1);

    for (int i = 0; i < N_LAYER; ++i) {
        conv(qkv_w   + (size_t)i * 3 * D_MODEL * D_MODEL, wqB,  (long long)3 * D_MODEL * D_MODEL);
        conv(o_w     + (size_t)i * D_MODEL * D_MODEL,     woB,  (long long)D_MODEL * D_MODEL);
        conv(r_net_w + (size_t)i * D_MODEL * D_MODEL,     wrB,  (long long)D_MODEL * D_MODEL);
        conv(ff1_w   + (size_t)i * D_FF * D_MODEL,        wf1B, (long long)D_FF * D_MODEL);
        conv(ff2_w   + (size_t)i * D_MODEL * D_FF,        wf2B, (long long)D_MODEL * D_FF);

        // LN(cat(mems=0, h)) -> bf16 [B*K, D]
        ln_kernel<<<BK, 256, 0, stream>>>(hF, ln1_g + i * D_MODEL, ln1_b + i * D_MODEL, XlnB, 1);

        // qkv = LN(cat) @ qkv_w^T : [B*K, 1536]
        gemm(XlnB, 0, 0, D_MODEL, wqB, 0, 0, D_MODEL,
             qkvC, nullptr, 0, 0, 3 * D_MODEL, nullptr, nullptr,
             1.0f, BK, 3 * D_MODEL, D_MODEL, 0, 1);
        split_qkv_kernel<<<BK, 256, 0, stream>>>(qkvC, r_w_bias, r_r_bias, QwB, QrB, KhB, VtB);

        // rk = pos_emb @ r_net_w^T -> [H, K, 64]
        gemm(posB, 0, 0, D_MODEL, wrB, 0, 0, D_MODEL,
             rkC, nullptr, 0, 0, D_MODEL, nullptr, nullptr,
             1.0f, KLEN, D_MODEL, D_MODEL, 0, 1);
        rk_permute_kernel<<<KLEN, 256, 0, stream>>>(rkC, rkB);

        // AC[b,h] = Qw @ Kh^T : batched [512,1024], Kd=64
        gemm(QwB, (long long)N_HEAD * QLEN * D_HEAD, (long long)QLEN * D_HEAD, D_HEAD,
             KhB, (long long)N_HEAD * KLEN * D_HEAD, (long long)KLEN * D_HEAD, D_HEAD,
             ACf, nullptr, (long long)N_HEAD * QLEN * KLEN, (long long)QLEN * KLEN, KLEN,
             nullptr, nullptr, 1.0f, QLEN, KLEN, D_HEAD, 0, BH);
        // BDraw[b,h] = Qr @ rk[h]^T (rk shared across batch: wS1=0)
        gemm(QrB, (long long)N_HEAD * QLEN * D_HEAD, (long long)QLEN * D_HEAD, D_HEAD,
             rkB, 0, (long long)KLEN * D_HEAD, D_HEAD,
             BDf, nullptr, (long long)N_HEAD * QLEN * KLEN, (long long)QLEN * KLEN, KLEN,
             nullptr, nullptr, 1.0f, QLEN, KLEN, D_HEAD, 0, BH);

        // rel-shift + mask + softmax -> prob bf16 [B,H,Q,K]
        attn_softmax_kernel<<<BH * QLEN, 256, 0, stream>>>(ACf, BDf, probB);

        // attn[b,h] = prob @ Vt^T -> written interleaved as [B,Q,H*64]
        gemm(probB, (long long)N_HEAD * QLEN * KLEN, (long long)QLEN * KLEN, KLEN,
             VtB,   (long long)N_HEAD * D_HEAD * KLEN, (long long)D_HEAD * KLEN, KLEN,
             nullptr, attnB, (long long)QLEN * D_MODEL, 64, D_MODEL,
             nullptr, nullptr, 1.0f, QLEN, D_HEAD, KLEN, 0, BH);

        // h += attn @ o_w^T
        gemm(attnB, 0, 0, D_MODEL, woB, 0, 0, D_MODEL,
             hF, nullptr, 0, 0, D_MODEL, nullptr, hF,
             1.0f, BQ, D_MODEL, D_MODEL, 0, 1);

        // FF block: h += relu(LN(h) @ ff1^T + b1) @ ff2^T + b2
        ln_kernel<<<BQ, 256, 0, stream>>>(hF, ln2_g + i * D_MODEL, ln2_b + i * D_MODEL, X2B, 0);
        gemm(X2B, 0, 0, D_MODEL, wf1B, 0, 0, D_MODEL,
             nullptr, XffB, 0, 0, D_FF, ff1_b + i * D_FF, nullptr,
             1.0f, BQ, D_FF, D_MODEL, 1, 1);
        gemm(XffB, 0, 0, D_FF, wf2B, 0, 0, D_FF,
             hF, nullptr, 0, 0, D_MODEL, ff2_b + i * D_MODEL, hF,
             1.0f, BQ, D_MODEL, D_FF, 0, 1);
    }

    // ---- vocab head: out = h @ head_w^T + head_b
    conv(hF, hB, (long long)BQ * D_MODEL);
    gemm(hB, 0, 0, D_MODEL, headB, 0, 0, D_MODEL,
         outp, nullptr, 0, 0, N_TOKEN, head_b, nullptr,
         1.0f, BQ, N_TOKEN, D_MODEL, 0, 1);
}
